// AttentionRNN_82145544503711
// MI455X (gfx1250) — matrix operand order repaired and checked by reading, not compiled, not run on hardware
//
#include <hip/hip_runtime.h>
#include <hip/hip_bf16.h>
#include <cstdint>

typedef __attribute__((ext_vector_type(16))) _Float16 v16h;
typedef __attribute__((ext_vector_type(8)))  _Float16 v8h;
typedef __attribute__((ext_vector_type(8)))  float    v8f;

#define NEGV (-10000000.0f)

// ---------------------------------------------------------------- helpers

__device__ __forceinline__ v8f wmma_f16(v16h a, v16h b, v8f c) {
  // D = A(16x32) * B(32x16) + C, f32 accumulate
  return __builtin_amdgcn_wmma_f32_16x16x32_f16(
      /*neg_a=*/false, a, /*neg_b=*/false, b,
      /*c_mod=*/(short)0, c, /*reuse_a=*/false, /*reuse_b=*/false);
}

// A-matrix 16x32 f16 lane layout (ISA 7.12.2): lane L holds row m=L&15,
// K in {kb..kb+7} U {kb+16..kb+23} with kb = (L<16 ? 0 : 8) + k0.
__device__ __forceinline__ v16h load_a(const _Float16* base, int ld, int m0,
                                       int k0, int lane) {
  int r  = m0 + (lane & 15);
  int kb = k0 + ((lane >> 4) << 3);
  const _Float16* p = base + (size_t)r * ld + kb;
  union { v16h v; v8h h[2]; } u;
  u.h[0] = *(const v8h*)(p);
  u.h[1] = *(const v8h*)(p + 16);
  return u.v;
}

// B-matrix 32x16 f16 lane layout: lane L holds col n=L&15,
// K in {kb..kb+7} U {kb+16..kb+23} with kb = (L<16 ? 0 : 8) + k0.  `base` is the
// column-major view of B, i.e. row n holds K contiguously (W row-major).
__device__ __forceinline__ v16h load_b(const _Float16* base, int ld, int n0,
                                       int k0, int lane) {
  int r  = n0 + (lane & 15);
  int kb = k0 + ((lane >> 4) << 3);
  const _Float16* p = base + (size_t)r * ld + kb;
  union { v16h v; v8h h[2]; } u;
  u.h[0] = *(const v8h*)(p);
  u.h[1] = *(const v8h*)(p + 16);
  return u.v;
}

__device__ __forceinline__ float wred(float v) {
  #pragma unroll
  for (int o = 16; o > 0; o >>= 1) v += __shfl_xor(v, o, 32);
  return v;
}
__device__ __forceinline__ float wmaxr(float v) {
  #pragma unroll
  for (int o = 16; o > 0; o >>= 1) v = fmaxf(v, __shfl_xor(v, o, 32));
  return v;
}

// ---------------------------------------------------------------- kernels

// cast f32 (rows x K) -> f16 (rows x Kpad), zero-pad K..Kpad
__global__ void k_pack_f16(const float* __restrict__ src,
                           _Float16* __restrict__ dst,
                           int rows, int K, int Kpad) {
  size_t total = (size_t)rows * Kpad;
  for (size_t i = (size_t)blockIdx.x * blockDim.x + threadIdx.x; i < total;
       i += (size_t)gridDim.x * blockDim.x) {
    int r = (int)(i / Kpad), k = (int)(i % Kpad);
    dst[i] = (k < K) ? (_Float16)src[(size_t)r * K + k] : (_Float16)0.f;
  }
}

// x16[row][k] = emb[tok[row]][k] (f16, K padded 300->320). emb[0]==0 so
// padding tokens produce zero rows.
__global__ void k_gather(const int* __restrict__ tok,
                         const float* __restrict__ emb,
                         _Float16* __restrict__ x16, int E, int Kpad) {
  int row = blockIdx.x;
  const float* e = emb + (size_t)tok[row] * E;
  for (int k = threadIdx.x; k < Kpad; k += blockDim.x)
    x16[(size_t)row * Kpad + k] = (k < E) ? (_Float16)e[k] : (_Float16)0.f;
}

// C[M x N] (f16) = X(f16,MxK) * W^T (W f16, NxK row-major) + bias.  N=1536.
// grid = (N/128, M/16), block = 256 (8 waves, one 16x16 tile each).
__global__ void __launch_bounds__(256)
k_gemm_xproj(const _Float16* __restrict__ X, const _Float16* __restrict__ W,
             const float* __restrict__ bias, _Float16* __restrict__ C,
             int N, int K) {
  int wave = threadIdx.x >> 5, lane = threadIdx.x & 31;
  int n0 = (blockIdx.x * 8 + wave) * 16;
  int m0 = blockIdx.y * 16;
  v8f c = {};
  for (int k0 = 0; k0 < K; k0 += 32) {
    if (k0 + 32 < K)
      __builtin_prefetch(W + (size_t)n0 * K + k0 + 32, 0, 1);
    v16h a = load_a(X, K, m0, k0, lane);
    v16h b = load_b(W, K, n0, k0, lane);
    c = wmma_f16(a, b, c);
  }
  int n  = n0 + (lane & 15);
  int mb = m0 + ((lane >> 4) << 3);
  float bv = bias[n];
  #pragma unroll
  for (int r = 0; r < 8; ++r)
    C[(size_t)(mb + r) * N + n] = (_Float16)(c[r] + bv);
}

// Bidirectional GRU scan; one block per direction (blockIdx.x).
// State h (64x256) lives in registers (32 f32/thread).  Per step:
//   1) fire async global->LDS copy of this step's xproj slab (64x768 f16)
//   2) cast h into LDS, WMMA HP = h @ Whh^T + bhh (192 tiles / 16 waves)
//   3) s_wait_asynccnt + barrier, then gate math from LDS.
// xproj (f16) layout: [b*T+t][dir*768 + 3H].
// LDS map (dynamic only, so offsets are LDS byte addresses):
//   [0,       32768)  h16   64x256 f16
//   [32768,  131072)  hp    64x768 f16
//   [131072, 229376)  xbuf  64x768 f16
__global__ void __launch_bounds__(512)
k_gru_scan(const int* __restrict__ tok, const _Float16* __restrict__ xproj,
           const _Float16* __restrict__ Whh16, const float* __restrict__ bhh,
           float* __restrict__ Hout, _Float16* __restrict__ H16out,
           int T, int Tpad) {
  extern __shared__ char smem[];
  _Float16* h16  = (_Float16*)smem;                    // 64*256
  _Float16* hp   = (_Float16*)(smem + 32768);          // 64*768
  _Float16* xbuf = (_Float16*)(smem + 131072);         // 64*768
  const int XBUF_LDS = 131072;
  const int dir = blockIdx.x;
  const int tid = threadIdx.x, wave = tid >> 5, lane = tid & 31;
  const _Float16* Wd = Whh16 + (size_t)dir * 768 * 256;
  const float* bd = bhh + dir * 768;
  float hreg[32];
  #pragma unroll
  for (int it = 0; it < 32; ++it) hreg[it] = 0.f;

  for (int t = 0; t < T; ++t) {
    const int tt = dir ? (T - 1 - t) : t;

    // Phase 0: async copy of xproj slab for timestep tt into LDS.
    // 64 rows x 768 halves = 6144 x b128, 12 per thread.
    #pragma unroll
    for (int i2 = 0; i2 < 12; ++i2) {
      int e = tid + i2 * 512;
      int m = e / 96, cch = e - m * 96;
      const _Float16* g =
          xproj + ((size_t)m * T + tt) * 1536 + dir * 768 + cch * 8;
      int lo = XBUF_LDS + (m * 768 + cch * 8) * 2;
      asm volatile("global_load_async_to_lds_b128 %0, %1, off"
                   :: "v"(lo), "v"(g) : "memory");
    }

    // Phase 1: publish h as f16 for the WMMA A operand.
    #pragma unroll
    for (int it = 0; it < 32; ++it) h16[tid + it * 512] = (_Float16)hreg[it];
    __syncthreads();

    // Phase 2: HP = h @ Whh^T + bhh (overlaps with the async copy).
    #pragma unroll 1
    for (int i = 0; i < 12; ++i) {   // 192 tiles / 16 waves
      int tIdx = wave * 12 + i;
      int m0 = (tIdx / 48) * 16, n0 = (tIdx % 48) * 16;
      v8f c = {};
      #pragma unroll
      for (int k0 = 0; k0 < 256; k0 += 32) {
        v16h a = load_a(h16, 256, m0, k0, lane);
        v16h b = load_b(Wd, 256, n0, k0, lane);
        c = wmma_f16(a, b, c);
      }
      int n  = n0 + (lane & 15);
      int mb = m0 + ((lane >> 4) << 3);
      float bv = bd[n];
      #pragma unroll
      for (int r = 0; r < 8; ++r)
        hp[(mb + r) * 768 + n] = (_Float16)(c[r] + bv);
    }
    asm volatile("s_wait_asynccnt 0x0" ::: "memory");
    __syncthreads();

    // Phase 3: gates.
    #pragma unroll 1
    for (int it = 0; it < 32; ++it) {
      int i = tid + it * 512;
      int m = i >> 8, j = i & 255;
      float mt = (tok[m * T + tt] != 0) ? 1.f : 0.f;
      const _Float16* xr = xbuf + m * 768;
      float hv  = hreg[it];
      float rg  = (float)xr[j]       + (float)hp[m * 768 + j];
      float zg  = (float)xr[256 + j] + (float)hp[m * 768 + 256 + j];
      float hpn = (float)hp[m * 768 + 512 + j];
      rg = 1.f / (1.f + __expf(-rg));
      zg = 1.f / (1.f + __expf(-zg));
      float ng = tanhf((float)xr[512 + j] + rg * hpn);
      float hn = (1.f - zg) * ng + zg * hv;
      hn = mt * hn + (1.f - mt) * hv;
      hreg[it] = hn;
      float o = hn * mt;
      Hout[((size_t)m * T + tt) * 512 + dir * 256 + j] = o;
      if (H16out)
        H16out[((size_t)m * Tpad + tt) * 512 + dir * 256 + j] = (_Float16)o;
    }
    __syncthreads();   // protects xbuf/hp/h16 from next step's writes
  }
}

// Build attention-side question operands from Hq32 (B,50,512):
//  Hq3[b][q][d] = Hq*w3_start (f16, q padded to 64)  -> B operand for scores
//  HqT[b][d][q] = Hq           (f16, transposed)     -> B operand for wavg
__global__ void k_build_q(const float* __restrict__ Hq32,
                          const float* __restrict__ sw,
                          _Float16* __restrict__ Hq3,
                          _Float16* __restrict__ HqT) {
  int b = blockIdx.x;
  for (int i = threadIdx.x; i < 64 * 512; i += blockDim.x) {
    int q = i >> 9, d = i & 511;
    float v = (q < 50) ? Hq32[((size_t)b * 50 + q) * 512 + d] : 0.f;
    Hq3[((size_t)b * 64 + q) * 512 + d] = (_Float16)(v * sw[1024 + d]);
    HqT[((size_t)b * 512 + d) * 64 + q] = (_Float16)v;
  }
}

// s1s/s1e[b][p] = Hp . w1_{start,end}.  grid (B, 50), 8 waves -> 8 rows.
__global__ void k_vec_p(const float* __restrict__ Hp32,
                        const float* __restrict__ sw,
                        const float* __restrict__ ew,
                        float* __restrict__ s1s, float* __restrict__ s1e) {
  int b = blockIdx.x, wave = threadIdx.x >> 5, lane = threadIdx.x & 31;
  int p = blockIdx.y * 8 + wave;
  const float* h = Hp32 + ((size_t)b * 400 + p) * 512;
  float as = 0.f, ae = 0.f;
  for (int d = lane; d < 512; d += 32) { float v = h[d]; as += v * sw[d]; ae += v * ew[d]; }
  as = wred(as); ae = wred(ae);
  if (lane == 0) { s1s[b * 400 + p] = as; s1e[b * 400 + p] = ae; }
}

// s2s[b][q] = Hq . w2_start
__global__ void k_vec_q(const float* __restrict__ Hq32,
                        const float* __restrict__ sw, float* __restrict__ s2s) {
  int b = blockIdx.x, q = threadIdx.x;
  if (q >= 50) return;
  const float* h = Hq32 + ((size_t)b * 50 + q) * 512;
  float a = 0.f;
  for (int d = 0; d < 512; ++d) a += h[d] * sw[512 + d];
  s2s[b * 50 + q] = a;
}

// Attention + logits.  grid (P/16, B), block 256 (8 waves).
__global__ void __launch_bounds__(256)
k_attn(const _Float16* __restrict__ Hp16, const float* __restrict__ Hp32,
       const _Float16* __restrict__ Hq3, const _Float16* __restrict__ HqT,
       const int* __restrict__ qtok, const int* __restrict__ ptok,
       const float* __restrict__ s1s, const float* __restrict__ s1e,
       const float* __restrict__ s2s,
       const float* __restrict__ sw, const float* __restrict__ ew,
       const float* __restrict__ sbp, const float* __restrict__ ebp,
       float* __restrict__ outS, float* __restrict__ outE) {
  __shared__ float    sG[16][64];
  __shared__ _Float16 sA[16][64];
  __shared__ float    sW[16][512];
  const int b = blockIdx.y, p0 = blockIdx.x * 16;
  const int wave = threadIdx.x >> 5, lane = threadIdx.x & 31;
  const float sb = sbp[0], eb = ebp[0];

  // Phase 1: G = Hp(16x512) @ (w3*Hq)^T (512x64) via WMMA, waves 0..3
  if (wave < 4) {
    const _Float16* Ap = Hp16 + (size_t)b * 400 * 512;
    const _Float16* Bq = Hq3 + (size_t)b * 64 * 512;
    int n0 = wave * 16;
    v8f c = {};
    for (int k0 = 0; k0 < 512; k0 += 32) {
      v16h a = load_a(Ap, 512, p0, k0, lane);
      v16h bm = load_b(Bq, 512, n0, k0, lane);
      c = wmma_f16(a, bm, c);
    }
    int n = n0 + (lane & 15), mb = (lane >> 4) << 3;
    #pragma unroll
    for (int r = 0; r < 8; ++r) sG[mb + r][n] = c[r];
  }
  __syncthreads();

  // Phase 2: masked softmax over valid q (equivalent to the reference's
  // score*mask -> softmax -> *mask -> renormalize).  2 rows per wave.
  for (int rr = wave * 2; rr < wave * 2 + 2; ++rr) {
    int p = p0 + rr;
    float base = s1s[b * 400 + p] + sb;
    float sc[2], ev[2], mx = -1e30f;
    #pragma unroll
    for (int h2 = 0; h2 < 2; ++h2) {
      int q = lane + h2 * 32;
      bool valid = (q < 50) && (qtok[b * 50 + q] != 0);
      float v = valid ? (base + s2s[b * 50 + q] + sG[rr][q]) : -1e30f;
      sc[h2] = v; mx = fmaxf(mx, v);
    }
    mx = wmaxr(mx);
    float sum = 0.f;
    #pragma unroll
    for (int h2 = 0; h2 < 2; ++h2) {
      ev[h2] = (sc[h2] > -1e29f) ? __expf(sc[h2] - mx) : 0.f;
      sum += ev[h2];
    }
    sum = wred(sum);
    float inv = 1.f / (sum + 1e-13f);
    #pragma unroll
    for (int h2 = 0; h2 < 2; ++h2)
      sA[rr][lane + h2 * 32] = (_Float16)(ev[h2] * inv);
  }
  __syncthreads();

  // Phase 3: wavg = a(16x64) @ Hq(64x512) via WMMA (HqT is K-contiguous)
  {
    const _Float16* Bm = HqT + (size_t)b * 512 * 64;
    #pragma unroll 1
    for (int i = 0; i < 4; ++i) {
      int n0 = (wave * 4 + i) * 16;
      v8f c = {};
      #pragma unroll
      for (int k0 = 0; k0 < 64; k0 += 32) {
        v16h a = load_a(&sA[0][0], 64, 0, k0, lane);
        v16h bm = load_b(Bm, 64, n0, k0, lane);
        c = wmma_f16(a, bm, c);
      }
      int n = n0 + (lane & 15), mb = (lane >> 4) << 3;
      #pragma unroll
      for (int r = 0; r < 8; ++r) sW[mb + r][n] = c[r];
    }
  }
  __syncthreads();

  // Phase 4: logits = Hp.w1 + wavg.w2 + (Hp*wavg).w3 + bias, masked by p.
  for (int rr = wave * 2; rr < wave * 2 + 2; ++rr) {
    int p = p0 + rr;
    const float* hrow = Hp32 + ((size_t)b * 400 + p) * 512;
    float a2s = 0.f, a3s = 0.f, a2e = 0.f, a3e = 0.f;
    for (int d = lane; d < 512; d += 32) {
      float wv = sW[rr][d], hv = hrow[d], pw = hv * wv;
      a2s += wv * sw[512 + d]; a3s += pw * sw[1024 + d];
      a2e += wv * ew[512 + d]; a3e += pw * ew[1024 + d];
    }
    a2s = wred(a2s); a3s = wred(a3s); a2e = wred(a2e); a3e = wred(a3e);
    if (lane == 0) {
      bool pm = ptok[b * 400 + p] != 0;
      outS[b * 400 + p] = pm ? (s1s[b * 400 + p] + a2s + a3s + sb) : NEGV;
      outE[b * 400 + p] = pm ? (s1e[b * 400 + p] + a2e + a3e + eb) : NEGV;
    }
  }
}

// log_softmax over P with log(p_mask + 1e-45) added.  One block per batch.
__global__ void __launch_bounds__(256)
k_lsm(const int* __restrict__ ptok, float* __restrict__ out) {
  const int b = blockIdx.x, tid = threadIdx.x;
  const int wave = tid >> 5, lane = tid & 31;
  __shared__ float red[8];
  const float LM = -103.6163f;  // logf(1e-45)
  for (int s = 0; s < 2; ++s) {
    const float* x = out + s * 25600 + b * 400;
    float* y = out + (2 + s) * 25600 + b * 400;
    float mx = -3.0e38f;
    for (int p = tid; p < 400; p += 256) {
      float z = x[p] + ((ptok[b * 400 + p] != 0) ? 0.f : LM);
      mx = fmaxf(mx, z);
    }
    mx = wmaxr(mx);
    if (lane == 0) red[wave] = mx;
    __syncthreads();
    float mAll = red[0];
    #pragma unroll
    for (int w = 1; w < 8; ++w) mAll = fmaxf(mAll, red[w]);
    __syncthreads();
    float sum = 0.f;
    for (int p = tid; p < 400; p += 256) {
      float z = x[p] + ((ptok[b * 400 + p] != 0) ? 0.f : LM);
      sum += __expf(z - mAll);
    }
    sum = wred(sum);
    if (lane == 0) red[wave] = sum;
    __syncthreads();
    float sAll = 0.f;
    #pragma unroll
    for (int w = 0; w < 8; ++w) sAll += red[w];
    float ls = __logf(sAll);
    __syncthreads();
    for (int p = tid; p < 400; p += 256) {
      float z = x[p] + ((ptok[b * 400 + p] != 0) ? 0.f : LM);
      y[p] = z - mAll - ls;
    }
    __syncthreads();
  }
}

// ---------------------------------------------------------------- launch

extern "C" void kernel_launch(void* const* d_in, const int* in_sizes, int n_in,
                              void* d_out, int out_size, void* d_ws,
                              size_t ws_size, hipStream_t stream) {
  (void)in_sizes; (void)n_in; (void)out_size; (void)ws_size;
  const int B = 64, P = 400, Q = 50, Qp = 64, E = 300, Kp = 320, N2 = 1536;

  const int*   passage = (const int*)d_in[0];
  const int*   question= (const int*)d_in[1];
  const float* emb     = (const float*)d_in[2];
  const float* pW_ih   = (const float*)d_in[3];
  const float* pW_hh   = (const float*)d_in[4];
  const float* pb_ih   = (const float*)d_in[5];
  const float* pb_hh   = (const float*)d_in[6];
  const float* qW_ih   = (const float*)d_in[7];
  const float* qW_hh   = (const float*)d_in[8];
  const float* qb_ih   = (const float*)d_in[9];
  const float* qb_hh   = (const float*)d_in[10];
  const float* start_w = (const float*)d_in[11];
  const float* start_b = (const float*)d_in[12];
  const float* end_w   = (const float*)d_in[13];
  const float* end_b   = (const float*)d_in[14];
  float* out = (float*)d_out;

  char* ws = (char*)d_ws;
  size_t off = 0;
  auto alloc = [&](size_t bytes) {
    size_t r = off; off = (off + bytes + 255) & ~(size_t)255; return r;
  };
  _Float16* pWih16 = (_Float16*)(ws + alloc((size_t)N2 * Kp * 2));
  _Float16* qWih16 = (_Float16*)(ws + alloc((size_t)N2 * Kp * 2));
  _Float16* pWhh16 = (_Float16*)(ws + alloc((size_t)N2 * 256 * 2));
  _Float16* qWhh16 = (_Float16*)(ws + alloc((size_t)N2 * 256 * 2));
  _Float16* x16p   = (_Float16*)(ws + alloc((size_t)B * P * Kp * 2));
  _Float16* x16q   = (_Float16*)(ws + alloc((size_t)B * Q * Kp * 2));
  _Float16* xpp    = (_Float16*)(ws + alloc((size_t)B * P * N2 * 2));
  _Float16* xpq    = (_Float16*)(ws + alloc((size_t)B * Q * N2 * 2));
  float*    Hp32   = (float*)(ws + alloc((size_t)B * P * 512 * 4));
  float*    Hq32   = (float*)(ws + alloc((size_t)B * Q * 512 * 4));
  _Float16* Hp16   = (_Float16*)(ws + alloc((size_t)B * P * 512 * 2));
  _Float16* Hq3    = (_Float16*)(ws + alloc((size_t)B * Qp * 512 * 2));
  _Float16* HqT    = (_Float16*)(ws + alloc((size_t)B * 512 * Qp * 2));
  float*    s1s    = (float*)(ws + alloc((size_t)B * P * 4));
  float*    s1e    = (float*)(ws + alloc((size_t)B * P * 4));
  float*    s2s    = (float*)(ws + alloc((size_t)B * Q * 4));

  // 1) weights -> f16 (ih padded K 300->320, hh K=256 exact)
  k_pack_f16<<<256, 256, 0, stream>>>(pW_ih, pWih16, N2, E, Kp);
  k_pack_f16<<<256, 256, 0, stream>>>(qW_ih, qWih16, N2, E, Kp);
  k_pack_f16<<<256, 256, 0, stream>>>(pW_hh, pWhh16, N2, 256, 256);
  k_pack_f16<<<256, 256, 0, stream>>>(qW_hh, qWhh16, N2, 256, 256);

  // 2) embedding gather -> f16
  k_gather<<<B * P, 256, 0, stream>>>(passage, emb, x16p, E, Kp);
  k_gather<<<B * Q, 256, 0, stream>>>(question, emb, x16q, E, Kp);

  // 3) input projections: x @ Wih^T + bih for both directions at once (f16 out)
  k_gemm_xproj<<<dim3(N2 / 128, (B * P) / 16), 256, 0, stream>>>(
      x16p, pWih16, pb_ih, xpp, N2, Kp);
  k_gemm_xproj<<<dim3(N2 / 128, (B * Q) / 16), 256, 0, stream>>>(
      x16q, qWih16, qb_ih, xpq, N2, Kp);

  // 4) GRU scans (one block per direction), 224 KB dynamic LDS
  const int SMEM = 32768 + 98304 + 98304;  // h16 + hp + xbuf = 229376 bytes
  (void)hipFuncSetAttribute((const void*)k_gru_scan,
                            hipFuncAttributeMaxDynamicSharedMemorySize, SMEM);
  k_gru_scan<<<2, 512, SMEM, stream>>>(passage, xpp, pWhh16, pb_hh,
                                       Hp32, Hp16, P, P);
  k_gru_scan<<<2, 512, SMEM, stream>>>(question, xpq, qWhh16, qb_hh,
                                       Hq32, (_Float16*)nullptr, Q, Qp);

  // 5) question-side attention operands
  k_build_q<<<B, 256, 0, stream>>>(Hq32, start_w, Hq3, HqT);

  // 6) per-row dot products
  k_vec_p<<<dim3(B, P / 8), 256, 0, stream>>>(Hp32, start_w, end_w, s1s, s1e);
  k_vec_q<<<B, 64, 0, stream>>>(Hq32, start_w, s2s);

  // 7) attention + logits (writes out[0], out[1])
  k_attn<<<dim3(P / 16, B), 256, 0, stream>>>(
      Hp16, Hp32, Hq3, HqT, question, passage, s1s, s1e, s2s,
      start_w, end_w, start_b, end_b, out, out + 25600);

  // 8) masked log-softmax over P (writes out[2], out[3])
  k_lsm<<<B, 256, 0, stream>>>(passage, out);
}
